// Mamba2Simple_4629974745535
// MI455X (gfx1250) — matrix operand order repaired and checked by reading, not compiled, not run on hardware
//
#include <hip/hip_runtime.h>
#include <hip/hip_bf16.h>

// ---------------------------------------------------------------------------
// Mamba2 forward for MI455X (gfx1250, wave32, WMMA).
// GEMMs: split-bf16 (hi+lo) 3-pass v_wmma_f32_16x16x32_bf16 (~fp32 accuracy).
// Tile feed: GLOBAL_LOAD_ASYNC_TO_LDS_B128 (ASYNCcnt) when the toolchain
// exposes the builtin, else classic global->VGPR->ds_store double buffering.
// ---------------------------------------------------------------------------

#if defined(__has_builtin)
#if __has_builtin(__builtin_amdgcn_global_load_async_to_lds_b128) && \
    __has_builtin(__builtin_amdgcn_global_load_async_to_lds_b32)  && \
    __has_builtin(__builtin_amdgcn_s_wait_asynccnt)
#define USE_ASYNC_LDS 1
#endif
#endif
#ifndef USE_ASYNC_LDS
#define USE_ASYNC_LDS 0
#endif

#define AS1 __attribute__((address_space(1)))
#define AS3 __attribute__((address_space(3)))

typedef int vint4 __attribute__((vector_size(16)));   // matches builtin param type

typedef __attribute__((ext_vector_type(16))) __bf16 v16bf;
typedef __attribute__((ext_vector_type(8)))  __bf16 v8bf;
typedef __attribute__((ext_vector_type(8)))  float  v8f;

union FragA { v16bf v; v8bf h[2]; };

constexpr int D_MODEL   = 1024;
constexpr int D_INNER   = 2048;
constexpr int D_STATE   = 64;
constexpr int HEADDIM   = 128;
constexpr int NHEADS    = 16;
constexpr int CONV_DIM  = 2176;   // D_INNER + 2*D_STATE
constexpr int D_IN_PROJ = 4240;   // 2*D_INNER + 2*D_STATE + NHEADS
constexpr int BATCH     = 2;
constexpr int SEQLEN    = 4096;
constexpr int MROWS     = BATCH * SEQLEN;  // 8192

// ---------------------------------------------------------------------------
// f32 -> (hi, lo) bf16 split
// ---------------------------------------------------------------------------
__global__ __launch_bounds__(256)
void cast_split(const float* __restrict__ src,
                __hip_bfloat16* __restrict__ hi,
                __hip_bfloat16* __restrict__ lo, int n)
{
    const int i = blockIdx.x * 256 + threadIdx.x;
    if (i >= n) return;
    const float v = src[i];
    const __hip_bfloat16 h = __float2bfloat16(v);
    hi[i] = h;
    lo[i] = __float2bfloat16(v - __bfloat162float(h));
}

// ---------------------------------------------------------------------------
// C[M,N] (f32) = A[M,K] x B[N,K]^T  with A,B given as bf16 hi/lo pairs.
// Block: 256 threads (8 waves), tile 128x128, TILE_K = 32, double-buffered LDS.
// Wave grid 2(M) x 4(N); each wave: 4x2 accumulators of 16x16.
// ---------------------------------------------------------------------------
__global__ __launch_bounds__(256)
void wmma_gemm_split(const __bf16* __restrict__ Ah, const __bf16* __restrict__ Al,
                     const __bf16* __restrict__ Bh, const __bf16* __restrict__ Bl,
                     float* __restrict__ C, int M, int N, int K)
{
    __shared__ __align__(64) __bf16 sAh[2][128][32];
    __shared__ __align__(64) __bf16 sAl[2][128][32];
    __shared__ __align__(64) __bf16 sBh[2][128][32];
    __shared__ __align__(64) __bf16 sBl[2][128][32];

    const int tid   = threadIdx.x;
    const int ldRow = tid >> 1;            // 0..127
    const int ldCol = (tid & 1) << 4;      // 0 or 16 elements
    const int mBase = blockIdx.x * 128;
    const int nBase = blockIdx.y * 128;

    const int wave = tid >> 5;
    const int lane = tid & 31;
    const int wm   = (wave & 1) << 6;      // 0 / 64
    const int wn   = (wave >> 1) << 5;     // 0 / 32 / 64 / 96
    const int half = lane >> 4;            // 0 / 1
    const int l16  = lane & 15;

    const v8f vzero = {0.f, 0.f, 0.f, 0.f, 0.f, 0.f, 0.f, 0.f};
    v8f acc[4][2];
#pragma unroll
    for (int mi = 0; mi < 4; ++mi)
#pragma unroll
        for (int ni = 0; ni < 2; ++ni) acc[mi][ni] = vzero;

#if USE_ASYNC_LDS
    // DMA global->LDS directly (ASYNCcnt); overlaps with WMMA on the prev tile.
    // OOB B rows (last N tile) are clamped: a garbage B *column* only affects
    // output columns gn >= N, which the store guard drops.
    auto issue_async = [&](int buf, int k0) {
        const size_t aoff = (size_t)(mBase + ldRow) * K + k0 + ldCol;
        __builtin_amdgcn_global_load_async_to_lds_b128(
            (AS1 vint4*)(Ah + aoff), (AS3 vint4*)&sAh[buf][ldRow][ldCol], 0, 0);
        __builtin_amdgcn_global_load_async_to_lds_b128(
            (AS1 vint4*)(Al + aoff), (AS3 vint4*)&sAl[buf][ldRow][ldCol], 0, 0);
        const int br = (nBase + ldRow < N) ? (nBase + ldRow) : (N - 1);
        const size_t boff = (size_t)br * K + k0 + ldCol;
        __builtin_amdgcn_global_load_async_to_lds_b128(
            (AS1 vint4*)(Bh + boff), (AS3 vint4*)&sBh[buf][ldRow][ldCol], 0, 0);
        __builtin_amdgcn_global_load_async_to_lds_b128(
            (AS1 vint4*)(Bl + boff), (AS3 vint4*)&sBl[buf][ldRow][ldCol], 0, 0);
    };

    issue_async(0, 0);
    __builtin_amdgcn_s_wait_asynccnt(0);
    __syncthreads();
#else
    auto gload = [&](int k0, uint4& ah, uint4& al, uint4& bh, uint4& bl) {
        const size_t aoff = (size_t)(mBase + ldRow) * K + k0 + ldCol;
        ah = *(const uint4*)(Ah + aoff);
        al = *(const uint4*)(Al + aoff);
        const int br = nBase + ldRow;
        if (br < N) {
            const size_t boff = (size_t)br * K + k0 + ldCol;
            bh = *(const uint4*)(Bh + boff);
            bl = *(const uint4*)(Bl + boff);
        } else {
            bh = make_uint4(0u, 0u, 0u, 0u);
            bl = make_uint4(0u, 0u, 0u, 0u);
        }
    };
    auto stage = [&](int buf, const uint4& ah, const uint4& al,
                     const uint4& bh, const uint4& bl) {
        *(uint4*)&sAh[buf][ldRow][ldCol] = ah;
        *(uint4*)&sAl[buf][ldRow][ldCol] = al;
        *(uint4*)&sBh[buf][ldRow][ldCol] = bh;
        *(uint4*)&sBl[buf][ldRow][ldCol] = bl;
    };

    uint4 ah, al, bh, bl;
    gload(0, ah, al, bh, bl);
    stage(0, ah, al, bh, bl);
    __syncthreads();
#endif

    const int nk = K >> 5;
    for (int kt = 0; kt < nk; ++kt) {
        const int cur = kt & 1;
#if USE_ASYNC_LDS
        if (kt + 1 < nk) issue_async(cur ^ 1, (kt + 1) << 5);
#else
        if (kt + 1 < nk) gload((kt + 1) << 5, ah, al, bh, bl);
#endif

        // B fragments: lane l16 = column n, contiguous 16 K-values per lane half
        v16bf bfh[2], bfl[2];
#pragma unroll
        for (int ni = 0; ni < 2; ++ni) {
            bfh[ni] = *(const v16bf*)&sBh[cur][wn + ni * 16 + l16][half << 4];
            bfl[ni] = *(const v16bf*)&sBl[cur][wn + ni * 16 + l16][half << 4];
        }
#pragma unroll
        for (int mi = 0; mi < 4; ++mi) {
            // A fragment: lane m = l16, contiguous K {half*16..+15}
            const __bf16* aph = &sAh[cur][wm + mi * 16 + l16][0];
            const __bf16* apl = &sAl[cur][wm + mi * 16 + l16][0];
            FragA fah, fal;
            fah.h[0] = *(const v8bf*)(aph + (half << 4));
            fah.h[1] = *(const v8bf*)(aph + 8 + (half << 4));
            fal.h[0] = *(const v8bf*)(apl + (half << 4));
            fal.h[1] = *(const v8bf*)(apl + 8 + (half << 4));
#pragma unroll
            for (int ni = 0; ni < 2; ++ni) {
                acc[mi][ni] = __builtin_amdgcn_wmma_f32_16x16x32_bf16(
                    false, fah.v, false, bfh[ni], (short)0, acc[mi][ni], false, false);
                acc[mi][ni] = __builtin_amdgcn_wmma_f32_16x16x32_bf16(
                    false, fah.v, false, bfl[ni], (short)0, acc[mi][ni], false, false);
                acc[mi][ni] = __builtin_amdgcn_wmma_f32_16x16x32_bf16(
                    false, fal.v, false, bfh[ni], (short)0, acc[mi][ni], false, false);
            }
        }
#if USE_ASYNC_LDS
        if (kt + 1 < nk) __builtin_amdgcn_s_wait_asynccnt(0);
#else
        if (kt + 1 < nk) stage(cur ^ 1, ah, al, bh, bl);
#endif
        __syncthreads();
    }

    // C/D layout: VGPR r -> M = half*8 + r, N = l16
#pragma unroll
    for (int mi = 0; mi < 4; ++mi) {
#pragma unroll
        for (int ni = 0; ni < 2; ++ni) {
            const int gn = nBase + wn + ni * 16 + l16;
            if (gn < N) {
#pragma unroll
                for (int r = 0; r < 8; ++r) {
                    const int gm = mBase + wm + mi * 16 + (half << 3) + r;
                    C[(size_t)gm * N + gn] = acc[mi][ni][r];
                }
            }
        }
    }
}

// ---------------------------------------------------------------------------
// Depthwise causal conv(4) + SiLU over xBC channels of zxbcdt
// ---------------------------------------------------------------------------
__global__ __launch_bounds__(256)
void conv_silu(const float* __restrict__ zx, const float* __restrict__ conv_w,
               const float* __restrict__ conv_b, float* __restrict__ xBC)
{
    const int idx = blockIdx.x * 256 + threadIdx.x;
    if (idx >= MROWS * CONV_DIM) return;
    const int c  = idx % CONV_DIM;
    const int bl = idx / CONV_DIM;
    const int l  = bl & (SEQLEN - 1);
    const float* src = zx + (size_t)bl * D_IN_PROJ + D_INNER + c;
    float acc = conv_b[c];
#pragma unroll
    for (int j = 0; j < 4; ++j) {
        const int lj = l - 3 + j;
        if (lj >= 0)
            acc = fmaf(conv_w[c * 4 + j], src[(long)(j - 3) * D_IN_PROJ], acc);
    }
    acc = acc / (1.f + __expf(-acc));   // SiLU
    xBC[(size_t)bl * CONV_DIM + c] = acc;
}

// ---------------------------------------------------------------------------
// dt = softplus(dt_raw + dt_bias)
// ---------------------------------------------------------------------------
__global__ __launch_bounds__(256)
void dt_softplus(const float* __restrict__ zx, const float* __restrict__ dt_bias,
                 float* __restrict__ dts)
{
    const int i = blockIdx.x * 256 + threadIdx.x;
    if (i >= MROWS * NHEADS) return;
    const int h  = i & (NHEADS - 1);
    const int bl = i >> 4;
    const float x = zx[(size_t)bl * D_IN_PROJ + (D_IN_PROJ - NHEADS) + h] + dt_bias[h];
    dts[i] = (x > 20.f) ? x : log1pf(__expf(x));
}

// ---------------------------------------------------------------------------
// Selective scan. Grid: BATCH*NHEADS*(HEADDIM/32) blocks of 128 threads.
// Each lane owns (p, 16 states over n); reduce y over 4 lanes via shfl_xor.
// B/C staged to LDS per step (async DMA when available).
// ---------------------------------------------------------------------------
__global__ __launch_bounds__(128)
void ssm_scan(const float* __restrict__ xBC, const float* __restrict__ dts,
              const float* __restrict__ A_log, const float* __restrict__ Dvec,
              float* __restrict__ y)
{
    const int chunk = blockIdx.x & 3;
    const int h     = (blockIdx.x >> 2) & (NHEADS - 1);
    const int b     = blockIdx.x >> 6;
    const int t     = threadIdx.x;
    const int p     = (chunk << 5) + (t >> 2);  // 0..127
    const int nq    = t & 3;

    const float A  = -__expf(A_log[h]);
    const float Dh = Dvec[h];

    __shared__ float sB[D_STATE];
    __shared__ float sC[D_STATE];
    __shared__ float sdt[1];

    const float* base = xBC + (size_t)b * SEQLEN * CONV_DIM;
    const float* dtb  = dts + (size_t)b * SEQLEN * NHEADS + h;
    float* yb = y + (size_t)b * SEQLEN * D_INNER + (size_t)h * HEADDIM;

    float st[16];
#pragma unroll
    for (int j = 0; j < 16; ++j) st[j] = 0.f;

    for (int ts = 0; ts < SEQLEN; ++ts) {
        const float* row = base + (size_t)ts * CONV_DIM;
#if USE_ASYNC_LDS
        if (t < 64)
            __builtin_amdgcn_global_load_async_to_lds_b32(
                (AS1 int*)(row + D_INNER + t), (AS3 int*)&sB[t], 0, 0);
        else
            __builtin_amdgcn_global_load_async_to_lds_b32(
                (AS1 int*)(row + D_INNER + D_STATE + (t - 64)),
                (AS3 int*)&sC[t - 64], 0, 0);
        if (t == 0) sdt[0] = dtb[(size_t)ts * NHEADS];
        __builtin_amdgcn_s_wait_asynccnt(0);
#else
        if (t < 64) sB[t] = row[D_INNER + t];
        else        sC[t - 64] = row[D_INNER + D_STATE + (t - 64)];
        if (t == 0) sdt[0] = dtb[(size_t)ts * NHEADS];
#endif
        __syncthreads();

        if (ts + 1 < SEQLEN)
            __builtin_prefetch(row + CONV_DIM + D_INNER, 0, 0);  // next step B/C

        const float xt    = row[(size_t)h * HEADDIM + p];
        const float decay = __expf(sdt[0] * A);
        float part = 0.f;
#pragma unroll
        for (int j = 0; j < 16; ++j) {
            const int n   = (nq << 4) + j;
            const float s = fmaf(decay, st[j], sB[n] * xt);
            st[j] = s;
            part  = fmaf(s, sC[n], part);
        }
        part += __shfl_xor(part, 1);
        part += __shfl_xor(part, 2);
        if (nq == 0) yb[(size_t)ts * D_INNER + p] = part + Dh * xt;
        __syncthreads();
    }
}

// ---------------------------------------------------------------------------
// RMS norm + sigmoid(z) gate, emit bf16 hi/lo for final WMMA GEMM
// ---------------------------------------------------------------------------
__global__ __launch_bounds__(256)
void rms_gate(const float* __restrict__ y, const float* __restrict__ zx,
              const float* __restrict__ rms_w,
              __hip_bfloat16* __restrict__ gh, __hip_bfloat16* __restrict__ gl)
{
    const int bl  = blockIdx.x;
    const int tid = threadIdx.x;
    const float* yr = y  + (size_t)bl * D_INNER;
    const float* zr = zx + (size_t)bl * D_IN_PROJ;  // z = first D_INNER cols

    float vals[8];
    float ss = 0.f;
#pragma unroll
    for (int j = 0; j < 8; ++j) {
        const float v = yr[tid + j * 256];
        vals[j] = v;
        ss = fmaf(v, v, ss);
    }
#pragma unroll
    for (int o = 16; o > 0; o >>= 1) ss += __shfl_xor(ss, o);
    __shared__ float red[8];
    if ((tid & 31) == 0) red[tid >> 5] = ss;
    __syncthreads();
    float tot = 0.f;
#pragma unroll
    for (int j = 0; j < 8; ++j) tot += red[j];
    const float scale = rsqrtf(tot * (1.f / (float)D_INNER) + 1e-5f);

#pragma unroll
    for (int j = 0; j < 8; ++j) {
        const int i = tid + j * 256;
        float v = vals[j] * scale * rms_w[i];
        const float z = zr[i];
        v = v / (1.f + __expf(-z));     // * sigmoid(z)
        const __hip_bfloat16 hv = __float2bfloat16(v);
        gh[(size_t)bl * D_INNER + i] = hv;
        gl[(size_t)bl * D_INNER + i] = __float2bfloat16(v - __bfloat162float(hv));
    }
}

// ---------------------------------------------------------------------------
extern "C" void kernel_launch(void* const* d_in, const int* in_sizes, int n_in,
                              void* d_out, int out_size, void* d_ws, size_t ws_size,
                              hipStream_t stream)
{
    (void)in_sizes; (void)n_in; (void)out_size; (void)ws_size;

    const float* u       = (const float*)d_in[0];
    const float* w_in    = (const float*)d_in[1];
    const float* conv_w  = (const float*)d_in[2];
    const float* conv_b  = (const float*)d_in[3];
    const float* dt_bias = (const float*)d_in[4];
    const float* A_log   = (const float*)d_in[5];
    const float* Dvec    = (const float*)d_in[6];
    const float* rms_w   = (const float*)d_in[7];
    const float* w_out   = (const float*)d_in[8];
    float* out = (float*)d_out;

    char* w = (char*)d_ws;
    auto carve = [&](size_t bytes) -> void* {
        void* p = (void*)w;
        w += (bytes + 255) & ~(size_t)255;
        return p;
    };

    __hip_bfloat16* uh  = (__hip_bfloat16*)carve((size_t)MROWS * D_MODEL * 2);
    __hip_bfloat16* ul  = (__hip_bfloat16*)carve((size_t)MROWS * D_MODEL * 2);
    __hip_bfloat16* wih = (__hip_bfloat16*)carve((size_t)D_IN_PROJ * D_MODEL * 2);
    __hip_bfloat16* wil = (__hip_bfloat16*)carve((size_t)D_IN_PROJ * D_MODEL * 2);
    __hip_bfloat16* woh = (__hip_bfloat16*)carve((size_t)D_MODEL * D_INNER * 2);
    __hip_bfloat16* wol = (__hip_bfloat16*)carve((size_t)D_MODEL * D_INNER * 2);
    float* zx   = (float*)carve((size_t)MROWS * D_IN_PROJ * 4);
    float* xBC  = (float*)carve((size_t)MROWS * CONV_DIM * 4);
    float* dts  = (float*)carve((size_t)MROWS * NHEADS * 4);
    float* ybuf = (float*)carve((size_t)MROWS * D_INNER * 4);
    __hip_bfloat16* gh = (__hip_bfloat16*)carve((size_t)MROWS * D_INNER * 2);
    __hip_bfloat16* gl = (__hip_bfloat16*)carve((size_t)MROWS * D_INNER * 2);

    const int nU  = MROWS * D_MODEL;
    const int nWi = D_IN_PROJ * D_MODEL;
    const int nWo = D_MODEL * D_INNER;
    cast_split<<<dim3((nU  + 255) / 256), 256, 0, stream>>>(u,     uh,  ul,  nU);
    cast_split<<<dim3((nWi + 255) / 256), 256, 0, stream>>>(w_in,  wih, wil, nWi);
    cast_split<<<dim3((nWo + 255) / 256), 256, 0, stream>>>(w_out, woh, wol, nWo);

    // zxbcdt = u @ in_proj_w^T   (M=8192, N=4240, K=1024)
    wmma_gemm_split<<<dim3(MROWS / 128, (D_IN_PROJ + 127) / 128), 256, 0, stream>>>(
        (const __bf16*)uh, (const __bf16*)ul,
        (const __bf16*)wih, (const __bf16*)wil,
        zx, MROWS, D_IN_PROJ, D_MODEL);

    conv_silu<<<dim3((MROWS * CONV_DIM + 255) / 256), 256, 0, stream>>>(
        zx, conv_w, conv_b, xBC);
    dt_softplus<<<dim3((MROWS * NHEADS + 255) / 256), 256, 0, stream>>>(
        zx, dt_bias, dts);

    ssm_scan<<<dim3(BATCH * NHEADS * (HEADDIM / 32)), 128, 0, stream>>>(
        xBC, dts, A_log, Dvec, ybuf);

    rms_gate<<<dim3(MROWS), 256, 0, stream>>>(ybuf, zx, rms_w, gh, gl);

    // out = y_gated @ out_w^T    (M=8192, N=1024, K=2048)
    wmma_gemm_split<<<dim3(MROWS / 128, D_MODEL / 128), 256, 0, stream>>>(
        (const __bf16*)gh, (const __bf16*)gl,
        (const __bf16*)woh, (const __bf16*)wol,
        out, MROWS, D_MODEL, D_INNER);
}
